// Global_CrossAttention_17145509446495
// MI455X (gfx1250) — matrix operand order repaired and checked by reading, not compiled, not run on hardware
//
#include <hip/hip_runtime.h>
#include <hip/hip_bf16.h>
#include <stdint.h>

#define B_    4
#define H_    8
#define N_    1024
#define D_    1024
#define CIN_  64
#define INV_SCALE 0.03125f   // 1/sqrt(1024)

typedef __attribute__((ext_vector_type(16))) __bf16 v16bf;
typedef __attribute__((ext_vector_type(8)))  float  v8f;

union ABPack  { v16bf v; uint32_t u[8]; uint4 q[2]; };
union AccPack { v8f   v; float   f[8]; };

// round-to-nearest-even f32->bf16 pair, merged with one v_perm_b32
__device__ __forceinline__ uint32_t pack_bf16(float lo, float hi) {
  uint32_t ul = __builtin_bit_cast(uint32_t, lo);
  uint32_t uh = __builtin_bit_cast(uint32_t, hi);
  ul += 0x7FFFu + ((ul >> 16) & 1u);
  uh += 0x7FFFu + ((uh >> 16) & 1u);
  return __builtin_amdgcn_perm(uh, ul, 0x07060302u); // {uh[31:16], ul[31:16]}
}

__device__ __forceinline__ v8f wmma_bf16(const ABPack& a, const ABPack& b, v8f c) {
  return __builtin_amdgcn_wmma_f32_16x16x32_bf16(false, a.v, false, b.v,
                                                 (short)0, c, false, false);
}

// -------------------------------------------------------------------------
// Kernel 0: convert Wq/Wk/Wv (f32 [8192][64]) to bf16 packed dwords.
// -------------------------------------------------------------------------
__global__ __launch_bounds__(256) void wcvt_kernel(
    const float* __restrict__ Wq, const float* __restrict__ Wk,
    const float* __restrict__ Wv, uint32_t* __restrict__ Wqb,
    uint32_t* __restrict__ Wkb, uint32_t* __restrict__ Wvb)
{
  int idx = blockIdx.x * 256 + threadIdx.x;     // 3 * 262144 dwords
  int which = idx >> 18;
  int off = idx & 0x3FFFF;
  const float2* src = (const float2*)(which == 0 ? Wq : (which == 1 ? Wk : Wv));
  uint32_t* dst = (which == 0 ? Wqb : (which == 1 ? Wkb : Wvb));
  float2 v = src[off];
  dst[off] = pack_bf16(v.x, v.y);
}

// -------------------------------------------------------------------------
// Kernel 1: Q = Xs^T Wq^T (bf16), KP = Ns^T Wk^T + pos_q (bf16)
// -------------------------------------------------------------------------
__global__ __launch_bounds__(256) void proj_qkp_kernel(
    const float* __restrict__ xs, const float* __restrict__ ns,
    const float* __restrict__ coord,
    const uint32_t* __restrict__ Wqb, const uint32_t* __restrict__ Wkb,
    const float* __restrict__ Wlq, const float* __restrict__ blq,
    uint16_t* __restrict__ Qws, uint16_t* __restrict__ KPws)
{
  __shared__ __align__(16) uint32_t XS[16 * 36];   // [n][c/2], stride 36 dwords
  __shared__ __align__(16) uint32_t NS[16 * 36];
  __shared__ float CO[16 * 4];
  __shared__ __align__(16) float STG[8 * 16 * 20]; // per-wave 16x20 staging

  const int t  = threadIdx.x;
  const uint32_t bx = blockIdx.x;
  const int b     = bx >> 12;
  const int ntile = (bx >> 6) & 63;
  const int og    = bx & 63;
  const int n0    = ntile << 4;

  #pragma unroll
  for (int i = 0; i < 2; ++i) {
    int idx = t + (i << 8);
    int cp = idx >> 4, n = idx & 15;
    const float* px = xs + (((size_t)b * CIN_ + 2 * cp) << 10) + n0 + n;
    const float* pn = ns + (((size_t)b * CIN_ + 2 * cp) << 10) + n0 + n;
    XS[n * 36 + cp] = pack_bf16(px[0], px[1024]);
    NS[n * 36 + cp] = pack_bf16(pn[0], pn[1024]);
  }
  if (t < 48) {
    int j = t >> 4, n = t & 15;
    CO[n * 4 + j] = coord[(((size_t)b * 3 + j) << 10) + n0 + n];
  }
  __syncthreads();

  const int w   = t >> 5;
  const int ln  = t & 31;
  const int col = ln & 15;
  const int hi  = ln >> 4;
  const int otile = (og << 7) + (w << 4);
  const int o   = otile + col;
  const int h   = otile >> 10, d0 = otile & 1023;
  const int arow = col * 36;
  float* stg = STG + w * 320;
  const int srow = ln >> 1, shalf = ln & 1;

  // ---------------- Q tile ----------------
  {
    AccPack acc = {};
    #pragma unroll
    for (int ks = 0; ks < 2; ++ks) {
      ABPack a, bb;
      int abase = arow + ks * 16 + (hi ? 8 : 0);
      a.q[0] = *(const uint4*)(XS + abase);
      a.q[1] = *(const uint4*)(XS + abase + 4);
      const uint32_t* wr = Wqb + (size_t)o * 32 + ks * 16 + (hi ? 8 : 0);
      bb.q[0] = *(const uint4*)(wr);
      bb.q[1] = *(const uint4*)(wr + 4);
      acc.v = wmma_bf16(a, bb, acc.v);
    }
    #pragma unroll
    for (int r = 0; r < 8; ++r)
      stg[(r + (hi ? 8 : 0)) * 20 + col] = acc.f[r];
    __syncthreads();
    const float* s = stg + srow * 20 + shalf * 8;
    uint4 pkt;
    pkt.x = pack_bf16(s[0], s[1]); pkt.y = pack_bf16(s[2], s[3]);
    pkt.z = pack_bf16(s[4], s[5]); pkt.w = pack_bf16(s[6], s[7]);
    uint32_t* qdst = (uint32_t*)Qws +
        (((((size_t)b * H_ + h) << 10) + n0 + srow) << 9) + (d0 >> 1) + shalf * 4;
    *(uint4*)qdst = pkt;
    __syncthreads();
  }

  // ---------------- KP tile (+ pos_q) ----------------
  {
    AccPack acc = {};
    #pragma unroll
    for (int ks = 0; ks < 2; ++ks) {
      ABPack a, bb;
      int abase = arow + ks * 16 + (hi ? 8 : 0);
      a.q[0] = *(const uint4*)(NS + abase);
      a.q[1] = *(const uint4*)(NS + abase + 4);
      const uint32_t* wr = Wkb + (size_t)o * 32 + ks * 16 + (hi ? 8 : 0);
      bb.q[0] = *(const uint4*)(wr);
      bb.q[1] = *(const uint4*)(wr + 4);
      acc.v = wmma_bf16(a, bb, acc.v);
    }
    float wl0 = Wlq[o * 3], wl1 = Wlq[o * 3 + 1], wl2 = Wlq[o * 3 + 2];
    float bq  = blq[o];
    #pragma unroll
    for (int r = 0; r < 8; ++r) {
      int m = r + (hi ? 8 : 0);
      float pos = CO[m * 4] * wl0 + CO[m * 4 + 1] * wl1 + CO[m * 4 + 2] * wl2 + bq;
      stg[m * 20 + col] = acc.f[r] + pos;
    }
    __syncthreads();
    const float* s = stg + srow * 20 + shalf * 8;
    uint4 pkt;
    pkt.x = pack_bf16(s[0], s[1]); pkt.y = pack_bf16(s[2], s[3]);
    pkt.z = pack_bf16(s[4], s[5]); pkt.w = pack_bf16(s[6], s[7]);
    uint32_t* kdst = (uint32_t*)KPws +
        (((((size_t)b * H_ + h) << 10) + n0 + srow) << 9) + (d0 >> 1) + shalf * 4;
    *(uint4*)kdst = pkt;
  }
}

// -------------------------------------------------------------------------
// Kernel 2. LDS layout (all dynamic, base offset 0):
//   E   f32[16][1028]  @ 0       (65792 B)
//   Qb  bf16[16][1032] @ 65792   (33024 B)
//   NSb u32 [16][36]   @ 98816   ( 2304 B)
//   CO  f32 [16][4]    @ 101120  (  256 B)
//   INV f32 [16]       @ 101376  (   64 B)
// -------------------------------------------------------------------------
#define E_OFF   0
#define QB_OFF  65792
#define NSB_OFF 98816
#define CO_OFF  101120
#define INV_OFF 101376

__device__ __forceinline__ void loadA(ABPack& a, const uint32_t* arow, int ks) {
  a.q[0] = *(const uint4*)(arow + ks * 16);
  a.q[1] = *(const uint4*)(arow + ks * 16 + 4);
}
__device__ __forceinline__ void loadB(ABPack& bb, const uint32_t* kprow, int ks) {
  bb.q[0] = *(const uint4*)(kprow + ks * 16);
  bb.q[1] = *(const uint4*)(kprow + ks * 16 + 4);
}

__global__ __launch_bounds__(256) void attn_out_kernel(
    const uint16_t* __restrict__ Qws, const uint16_t* __restrict__ KPws,
    const float* __restrict__ ns, const float* __restrict__ coord,
    const uint32_t* __restrict__ Wvb, const float* __restrict__ Wlv,
    const float* __restrict__ blv, float* __restrict__ out)
{
  extern __shared__ char smem[];
  float*    E   = (float*)(smem + E_OFF);
  uint16_t* Qb  = (uint16_t*)(smem + QB_OFF);
  uint32_t* NSb = (uint32_t*)(smem + NSB_OFF);
  float*    CO  = (float*)(smem + CO_OFF);
  float*    INV = (float*)(smem + INV_OFF);

  const int t  = threadIdx.x;
  const uint32_t bx = blockIdx.x;
  const int bh = bx >> 6, ntile = bx & 63;
  const int b = bh >> 3, h = bh & 7;
  const int n0 = ntile << 4;

  // --- stage Q block via async DMA to LDS (16 x 512 dwords, padded 516) ---
  {
    const uint32_t* qsrc =
        (const uint32_t*)Qws + (((((size_t)b * H_ + h) << 10) + n0) << 9);
    #pragma unroll
    for (int i = 0; i < 8; ++i) {
      int idx = t + (i << 8);
      int r = idx >> 7, c4 = (idx & 127) << 2;
      uint64_t ga = (uint64_t)(uintptr_t)(qsrc + (r << 9) + c4);
      uint32_t la = QB_OFF + (uint32_t)(r * 516 + c4) * 4u;
      asm volatile("global_load_async_to_lds_b128 %0, %1, off"
                   :: "v"(la), "v"(ga) : "memory");
    }
  }
  // --- stage Ns block packed [n][c/2] and coords (regular path) ---
  #pragma unroll
  for (int i = 0; i < 2; ++i) {
    int idx = t + (i << 8);
    int cp = idx >> 4, n = idx & 15;
    const float* pn = ns + (((size_t)b * CIN_ + 2 * cp) << 10) + n0 + n;
    NSb[n * 36 + cp] = pack_bf16(pn[0], pn[1024]);
  }
  if (t < 48) {
    int j = t >> 4, n = t & 15;
    CO[n * 4 + j] = coord[(((size_t)b * 3 + j) << 10) + n0 + n];
  }
  asm volatile("s_wait_asynccnt 0x0" ::: "memory");
  __syncthreads();

  const int w   = t >> 5;
  const int ln  = t & 31;
  const int col = ln & 15;
  const int hi  = ln >> 4;

  // ------- phase 1: energy, two m-tiles per wave (independent acc chains) ---
  {
    const uint32_t* KPd =
        (const uint32_t*)KPws + ((((size_t)b * H_ + h) << 10) << 9);
    const uint32_t* arow = (const uint32_t*)Qb + col * 516 + (hi ? 8 : 0);
    for (int i = 0; i < 4; ++i) {
      int mt0 = (w << 1) + (i << 4);
      const uint32_t* kprow0 = KPd + (((mt0 << 4) + col) << 9) + (hi ? 8 : 0);
      const uint32_t* kprow1 = kprow0 + (16 << 9);
      AccPack acc0 = {}, acc1 = {};
      ABPack a0, a1, p0, p1, q0, q1;
      loadA(a0, arow, 0); loadB(p0, kprow0, 0); loadB(q0, kprow1, 0);
      #pragma unroll
      for (int ks = 0; ks < 30; ks += 2) {
        loadA(a1, arow, ks + 1); loadB(p1, kprow0, ks + 1); loadB(q1, kprow1, ks + 1);
        acc0.v = wmma_bf16(a0, p0, acc0.v);
        acc1.v = wmma_bf16(a0, q0, acc1.v);
        loadA(a0, arow, ks + 2); loadB(p0, kprow0, ks + 2); loadB(q0, kprow1, ks + 2);
        acc0.v = wmma_bf16(a1, p1, acc0.v);
        acc1.v = wmma_bf16(a1, q1, acc1.v);
      }
      loadA(a1, arow, 31); loadB(p1, kprow0, 31); loadB(q1, kprow1, 31);
      acc0.v = wmma_bf16(a0, p0, acc0.v);
      acc1.v = wmma_bf16(a0, q0, acc1.v);
      acc0.v = wmma_bf16(a1, p1, acc0.v);
      acc1.v = wmma_bf16(a1, q1, acc1.v);
      #pragma unroll
      for (int r = 0; r < 8; ++r) {
        int erow = (r + (hi ? 8 : 0)) * 1028 + (mt0 << 4) + col;
        E[erow]      = acc0.f[r] * INV_SCALE;
        E[erow + 16] = acc1.f[r] * INV_SCALE;
      }
    }
  }
  __syncthreads();

  // ---------------- phase 2: row softmax ----------------
  {
    int row = t >> 4, sub = t & 15;
    float* Er = E + row * 1028;
    float mx = -1e30f;
    for (int i = 0; i < 64; ++i) mx = fmaxf(mx, Er[sub + (i << 4)]);
    for (int off = 8; off; off >>= 1) mx = fmaxf(mx, __shfl_xor(mx, off, 16));
    float s = 0.f;
    for (int i = 0; i < 64; ++i) {
      float e = __expf(Er[sub + (i << 4)] - mx);
      Er[sub + (i << 4)] = e;
      s += e;
    }
    for (int off = 8; off; off >>= 1) s += __shfl_xor(s, off, 16);
    if (sub == 0) INV[row] = 1.f / s;
  }
  __syncthreads();

  // ------- phase 3: VC tiles (two d-tiles, interleaved) + attn multiply ----
  {
    for (int i = 0; i < 4; ++i) {
      int dt0 = (w << 1) + (i << 4);
      int o0 = (h << 10) + (dt0 << 4) + col;
      int o1 = o0 + 16;
      AccPack acc0 = {}, acc1 = {};
      #pragma unroll
      for (int ks = 0; ks < 2; ++ks) {
        ABPack a, b0, b1;
        int abase = col * 36 + ks * 16 + (hi ? 8 : 0);
        a.q[0] = *(const uint4*)(NSb + abase);
        a.q[1] = *(const uint4*)(NSb + abase + 4);
        const uint32_t* wr0 = Wvb + (size_t)o0 * 32 + ks * 16 + (hi ? 8 : 0);
        const uint32_t* wr1 = Wvb + (size_t)o1 * 32 + ks * 16 + (hi ? 8 : 0);
        b0.q[0] = *(const uint4*)(wr0); b0.q[1] = *(const uint4*)(wr0 + 4);
        b1.q[0] = *(const uint4*)(wr1); b1.q[1] = *(const uint4*)(wr1 + 4);
        acc0.v = wmma_bf16(a, b0, acc0.v);
        acc1.v = wmma_bf16(a, b1, acc1.v);
      }
      float wl00 = Wlv[o0 * 3], wl01 = Wlv[o0 * 3 + 1], wl02 = Wlv[o0 * 3 + 2];
      float wl10 = Wlv[o1 * 3], wl11 = Wlv[o1 * 3 + 1], wl12 = Wlv[o1 * 3 + 2];
      float bv0 = blv[o0], bv1 = blv[o1];
      #pragma unroll
      for (int r = 0; r < 8; ++r) {
        int n = r + (hi ? 8 : 0);
        float c0 = CO[n * 4], c1 = CO[n * 4 + 1], c2 = CO[n * 4 + 2];
        float iv = INV[n];
        int erow = n * 1028 + (dt0 << 4) + col;
        float pos0 = c0 * wl00 + c1 * wl01 + c2 * wl02 + bv0;
        float pos1 = c0 * wl10 + c1 * wl11 + c2 * wl12 + bv1;
        E[erow]      = E[erow]      * iv * (acc0.f[r] + pos0);
        E[erow + 16] = E[erow + 16] * iv * (acc1.f[r] + pos1);
      }
    }
  }
  __syncthreads();

  // ---- phase 4: transposed store, LDS -> global via async store DMA ----
  {
    int n = t & 15, dg = t >> 4;
    size_t obase = ((((size_t)b << 13) + (h << 10)) << 10);
    const float* orow = out + obase + n0 + n;
    for (int j = 0; j < 64; ++j) {
      int d = (dg << 6) + j;
      uint64_t ga = (uint64_t)(uintptr_t)(orow + ((size_t)d << 10));
      uint32_t la = E_OFF + (uint32_t)(n * 1028 + d) * 4u;
      asm volatile("global_store_async_from_lds_b32 %0, %1, off"
                   :: "v"(ga), "v"(la) : "memory");
    }
  }
}

extern "C" void kernel_launch(void* const* d_in, const int* in_sizes, int n_in,
                              void* d_out, int out_size, void* d_ws, size_t ws_size,
                              hipStream_t stream) {
  (void)in_sizes; (void)n_in; (void)out_size; (void)ws_size;
  const float* coord = (const float*)d_in[0];
  const float* x     = (const float*)d_in[1];
  const float* nb    = (const float*)d_in[2];
  const float* Wq    = (const float*)d_in[3];
  const float* Wk    = (const float*)d_in[4];
  const float* Wv    = (const float*)d_in[5];
  const float* Wlq   = (const float*)d_in[6];
  const float* blq   = (const float*)d_in[7];
  const float* Wlv   = (const float*)d_in[8];
  const float* blv   = (const float*)d_in[9];
  float* out = (float*)d_out;

  const size_t QK_ELEMS = (size_t)B_ * H_ * N_ * D_;
  uint16_t* Qws  = (uint16_t*)d_ws;
  uint16_t* KPws = Qws + QK_ELEMS;
  uint32_t* Wqb  = (uint32_t*)(KPws + QK_ELEMS);
  uint32_t* Wkb  = Wqb + 8192 * 32;
  uint32_t* Wvb  = Wkb + 8192 * 32;

  wcvt_kernel<<<3072, 256, 0, stream>>>(Wq, Wk, Wv, Wqb, Wkb, Wvb);

  proj_qkp_kernel<<<B_ * 64 * 64, 256, 0, stream>>>(
      x, nb, coord, Wqb, Wkb, Wlq, blq, Qws, KPws);

  size_t smem = 101376 + 64;                       // ~99.1 KB dynamic LDS
  attn_out_kernel<<<B_ * H_ * (N_ / 16), 256, smem, stream>>>(
      Qws, KPws, nb, coord, Wvb, Wlv, blv, out);
}
